// ConsciousnessIntegrator_61993557950908
// MI455X (gfx1250) — matrix operand order repaired and checked by reading, not compiled, not run on hardware
//
#include <hip/hip_runtime.h>
#include <cstdint>
#include <cstddef>

// ---------------------------------------------------------------------------
// ConsciousnessIntegrator forward for MI455X (gfx1250, wave32, WMMA bf16)
// GEMMs: LDS double-buffered tiles fed by GLOBAL_LOAD_ASYNC_TO_LDS_B128
// (ASYNCcnt pipeline) into v_wmma_f32_16x16x32_bf16. Glue ops on VALU.
// ---------------------------------------------------------------------------

typedef __attribute__((ext_vector_type(16))) __bf16 v16bf;
typedef __attribute__((ext_vector_type(8)))  float  v8f;
typedef __attribute__((ext_vector_type(4)))  int    v4i;

#define BATCH   2048
#define HDIM    1024
#define SEQ     16
#define TOKENS  (BATCH * SEQ)          // 32768
#define FFDIM   4096
#define NHEADS  8
#define HEADD   128
#define QKVN    (3 * HDIM)             // fused q|k|v output width

#define GLOBAL_AS __attribute__((address_space(1)))
#define LDS_AS    __attribute__((address_space(3)))

#if defined(__AMDGCN__) && \
    __has_builtin(__builtin_amdgcn_global_load_async_to_lds_b128) && \
    __has_builtin(__builtin_amdgcn_s_wait_asynccnt)
#define USE_ASYNC_LDS 1
#else
#define USE_ASYNC_LDS 0
#endif

// ---------------- device helpers ----------------

__device__ __forceinline__ float gelu_exact(float x) {
    return 0.5f * x * (1.0f + erff(x * 0.7071067811865475f));
}
__device__ __forceinline__ float sigmoidf(float x) {
    return 1.0f / (1.0f + expf(-x));
}

__device__ __forceinline__ float block_reduce_sum(float v, float* sdata) {
    int tid = threadIdx.x;
    sdata[tid] = v;
    __syncthreads();
    for (int s = 128; s > 0; s >>= 1) {
        if (tid < s) sdata[tid] += sdata[tid + s];
        __syncthreads();
    }
    float r = sdata[0];
    __syncthreads();
    return r;
}

// A fragment (16x32 bf16, MxK): lanes 0-15 hold K=[0..7]+[16..23],
// lanes 16-31 hold K=[8..15]+[24..31]  -> two 16B chunks 32B apart.
__device__ __forceinline__ v16bf load_frag_a(const __bf16* p) {
    union { uint4 u[2]; v16bf v; } t;
    t.u[0] = *reinterpret_cast<const uint4*>(p);
    t.u[1] = *reinterpret_cast<const uint4*>(p + 16);
    return t.v;
}
// B fragment (32x16 bf16, KxN): lanes 0-15 hold K=[0..7]+[16..23], lanes 16-31
// K=[8..15]+[24..31] for column n = lane&15 -> two 16B chunks 32B apart.
__device__ __forceinline__ v16bf load_frag_b(const __bf16* p) {
    union { uint4 u[2]; v16bf v; } t;
    t.u[0] = *reinterpret_cast<const uint4*>(p);
    t.u[1] = *reinterpret_cast<const uint4*>(p + 16);
    return t.v;
}

// ---------------- f32 -> bf16 conversion ----------------

__global__ __launch_bounds__(256) void f2bf_kernel(const float* __restrict__ in,
                                                   __bf16* __restrict__ out, size_t n) {
    size_t i = (size_t)blockIdx.x * 256 + threadIdx.x;
    if (i < n) out[i] = (__bf16)in[i];
}

// ---------------- bias packing for fused QKV ----------------

__global__ __launch_bounds__(256) void pack3_kernel(
    const float* __restrict__ a, const float* __restrict__ b,
    const float* __restrict__ c, float* __restrict__ out) {
    int i = blockIdx.x * 256 + threadIdx.x;
    if (i < HDIM)            out[i] = a[i];
    else if (i < 2 * HDIM)   out[i] = b[i - HDIM];
    else if (i < 3 * HDIM)   out[i] = c[i - 2 * HDIM];
}

// ---------------- WMMA bf16 GEMM: C[M,N] = A[M,K] @ W[N,K]^T + bias ----------------
// Block: 256 threads = 8 waves (2 M-waves x 4 N-waves). Wave tile 32x64,
// block tile 64x256. K-tiles (32) double-buffered in LDS, staged with
// async global->LDS DMA when available. act: 0 none, 1 relu.

__global__ __launch_bounds__(256) void gemm_bf16_wmma(
    const __bf16* __restrict__ A, const __bf16* __restrict__ W,
    const float* __restrict__ bias,
    float* __restrict__ outF, __bf16* __restrict__ outB,
    int M, int N, int K, int act)
{
    constexpr int KT = 32;            // k tile
    constexpr int LDK = 40;           // padded LDS row stride (elems)
    __shared__ __bf16 As[2][64 * LDK];    //  2 x  5 KB
    __shared__ __bf16 Ws[2][256 * LDK];   //  2 x 20 KB

    const int tid  = threadIdx.x;
    const int wave = tid >> 5;
    const int lane = tid & 31;
    const int wm   = wave >> 2;          // 0..1
    const int wn   = wave & 3;           // 0..3
    const int half = lane >> 4;
    const int lrow = lane & 15;

    const int mBlk = blockIdx.y * 64;
    const int nBlk = blockIdx.x * 256;

    // staging indices: one b128/thread for A (64x32), four for W (256x32)
    const int crow = tid >> 2;                 // 0..63
    const int cchk = (tid & 3) * 8;            // elem offset within row
    const __bf16* Ag = A + (size_t)(mBlk + crow) * K + cchk;
    const __bf16* Wg = W + (size_t)(nBlk + crow) * K + cchk;
    const size_t wstep = (size_t)64 * K;
    const int ldsOff = crow * LDK + cchk;

    auto stage = [&](int buf, int k) {
#if USE_ASYNC_LDS
        __builtin_amdgcn_global_load_async_to_lds_b128(
            (GLOBAL_AS v4i*)(Ag + k),
            (LDS_AS v4i*)&As[buf][ldsOff], 0, 0);
#pragma unroll
        for (int c = 0; c < 4; ++c)
            __builtin_amdgcn_global_load_async_to_lds_b128(
                (GLOBAL_AS v4i*)(Wg + k + wstep * c),
                (LDS_AS v4i*)&Ws[buf][ldsOff + c * 64 * LDK], 0, 0);
#else
        *(uint4*)&As[buf][ldsOff] = *(const uint4*)(Ag + k);
#pragma unroll
        for (int c = 0; c < 4; ++c)
            *(uint4*)&Ws[buf][ldsOff + c * 64 * LDK] =
                *(const uint4*)(Wg + k + wstep * c);
#endif
    };

    const v8f zero = {0.f, 0.f, 0.f, 0.f, 0.f, 0.f, 0.f, 0.f};
    v8f acc[2][4];
#pragma unroll
    for (int i = 0; i < 2; ++i)
#pragma unroll
        for (int j = 0; j < 4; ++j) acc[i][j] = zero;

    stage(0, 0);
    int buf = 0;
    for (int k = 0; k < K; k += KT) {
        if (k + KT < K) {
            stage(buf ^ 1, k + KT);            // prefetch next tile
#if USE_ASYNC_LDS
            __builtin_amdgcn_s_wait_asynccnt(5);   // retire previous batch (5 new in flight)
#endif
        } else {
#if USE_ASYNC_LDS
            __builtin_amdgcn_s_wait_asynccnt(0);
#endif
        }
        __syncthreads();

        const __bf16* a0p = &As[buf][(wm * 32 + lrow) * LDK + half * 8];
        v16bf a0 = load_frag_a(a0p);
        v16bf a1 = load_frag_a(a0p + 16 * LDK);
#pragma unroll
        for (int j = 0; j < 4; ++j) {
            v16bf bfr = load_frag_b(&Ws[buf][(wn * 64 + j * 16 + lrow) * LDK + half * 8]);
            acc[0][j] = __builtin_amdgcn_wmma_f32_16x16x32_bf16(
                false, a0, false, bfr, (short)0, acc[0][j], false, false);
            acc[1][j] = __builtin_amdgcn_wmma_f32_16x16x32_bf16(
                false, a1, false, bfr, (short)0, acc[1][j], false, false);
        }
        __syncthreads();
        buf ^= 1;
    }

    // Epilogue. C/D layout: VGPR r -> M = r + 8*half (within 16-row tile), N = lrow.
#pragma unroll
    for (int mi = 0; mi < 2; ++mi) {
#pragma unroll
        for (int j = 0; j < 4; ++j) {
            const int col = nBlk + wn * 64 + j * 16 + lrow;
            const float bv = bias ? bias[col] : 0.0f;
#pragma unroll
            for (int r = 0; r < 8; ++r) {
                const int row = mBlk + wm * 32 + mi * 16 + half * 8 + r;
                float v = acc[mi][j][r] + bv;
                if (act == 1) v = v > 0.0f ? v : 0.0f;
                const size_t idx = (size_t)row * N + col;
                if (outF) outF[idx] = v;
                if (outB) outB[idx] = (__bf16)v;
            }
        }
    }
}

// ---------------- projections: signals[b, slot, :] = x[b] @ Wt + bias ----------------

__global__ __launch_bounds__(256) void proj_kernel(
    const float* __restrict__ X, const float* __restrict__ W,
    const float* __restrict__ bias, float* __restrict__ signals,
    int d, int slot)
{
    const int b = blockIdx.x;
    const float* x = X + (size_t)b * d;
    float* out = signals + ((size_t)b * 4 + slot) * HDIM;
    for (int j = threadIdx.x; j < HDIM; j += 256) {
        const float* w = W + (size_t)j * d;
        float s = bias[j];
        for (int k = 0; k < d; ++k) s += x[k] * w[k];
        out[j] = s;
    }
}

// ---------------- salience MLP: sigmoid(W2 @ gelu(W1 x + b1) + b2) ----------------

__global__ __launch_bounds__(256) void sal_mlp_kernel(
    const float* __restrict__ X, const float* __restrict__ W1,
    const float* __restrict__ B1, const float* __restrict__ W2,
    const float* __restrict__ B2, float* __restrict__ sal, int d, int slot)
{
    __shared__ float sdata[256];
    const int b = blockIdx.x;
    const float* x = X + (size_t)b * d;
    float part = 0.0f;
    for (int i = threadIdx.x; i < 512; i += 256) {
        const float* w = W1 + (size_t)i * d;
        float s = B1[i];
        for (int k = 0; k < d; ++k) s += x[k] * w[k];
        part += gelu_exact(s) * W2[i];
    }
    float tot = block_reduce_sum(part, sdata);
    if (threadIdx.x == 0) sal[(size_t)b * 4 + slot] = sigmoidf(tot + B2[0]);
}

__global__ __launch_bounds__(256) void sal_agency_kernel(
    const float* __restrict__ X, const float* __restrict__ W,
    float* __restrict__ sal, int n)
{
    int b = blockIdx.x * 256 + threadIdx.x;
    if (b < n) sal[(size_t)b * 4 + 2] = W[0] * X[b];   // Linear(1,1,bias=False), no sigmoid
}

// ---------------- stable top-k(4) + workspace build ----------------

__global__ __launch_bounds__(256) void build_ws_kernel(
    const float* __restrict__ signals, const float* __restrict__ sal,
    float* __restrict__ X, __bf16* __restrict__ XB)
{
    const int b = blockIdx.x;
    __shared__ int order[4];
    if (threadIdx.x == 0) {
        float s[4];
        bool used[4] = {false, false, false, false};
        for (int i = 0; i < 4; ++i) s[i] = sal[(size_t)b * 4 + i];
        for (int r = 0; r < 4; ++r) {
            int best = -1; float bv = 0.0f;
            for (int i = 0; i < 4; ++i)
                if (!used[i] && (best < 0 || s[i] > bv)) { best = i; bv = s[i]; }
            used[best] = true;
            order[r] = best;       // strict '>' with ascending scan == stable top_k
        }
    }
    __syncthreads();
    const size_t xb = (size_t)b * SEQ * HDIM;
    for (int idx = threadIdx.x; idx < SEQ * HDIM; idx += 256) {
        const int t = idx >> 10, hcol = idx & 1023;
        float v = (t < 4) ? signals[((size_t)b * 4 + order[t]) * HDIM + hcol] : 0.0f;
        X[xb + idx]  = v;
        XB[xb + idx] = (__bf16)v;
    }
}

// ---------------- attention over fused QKV [B*16, 3072]: seq 16, hd 128 ----------------

__global__ __launch_bounds__(256) void attention_kernel(
    const float* __restrict__ QKV, __bf16* __restrict__ O)
{
    const int b = blockIdx.x, h = blockIdx.y;
    __shared__ float sq[SEQ][HEADD], sk[SEQ][HEADD], sv[SEQ][HEADD];
    __shared__ float sp[SEQ][SEQ], rmax[SEQ], rsum[SEQ];

    const size_t rb = (size_t)b * SEQ * QKVN + (size_t)h * HEADD;
    for (int i = threadIdx.x; i < SEQ * HEADD; i += 256) {
        const int t = i >> 7, d = i & 127;
        const size_t p = rb + (size_t)t * QKVN + d;
        sq[t][d] = QKV[p];
        sk[t][d] = QKV[p + HDIM];
        sv[t][d] = QKV[p + 2 * HDIM];
    }
    __syncthreads();

    const int i = threadIdx.x >> 4, j = threadIdx.x & 15;
    float s = 0.0f;
    for (int d = 0; d < HEADD; ++d) s += sq[i][d] * sk[j][d];
    sp[i][j] = s * 0.08838834764831845f;   // 1/sqrt(128)
    __syncthreads();

    if (j == 0) {
        float mx = sp[i][0];
        for (int jj = 1; jj < SEQ; ++jj) mx = fmaxf(mx, sp[i][jj]);
        float sm = 0.0f;
        for (int jj = 0; jj < SEQ; ++jj) sm += expf(sp[i][jj] - mx);
        rmax[i] = mx; rsum[i] = sm;
    }
    __syncthreads();
    sp[i][j] = expf(sp[i][j] - rmax[i]) / rsum[i];
    __syncthreads();

    for (int idx = threadIdx.x; idx < SEQ * HEADD; idx += 256) {
        const int t = idx >> 7, d = idx & 127;
        float o = 0.0f;
        for (int jj = 0; jj < SEQ; ++jj) o += sp[t][jj] * sv[jj][d];
        O[((size_t)b * SEQ + t) * HDIM + (size_t)h * HEADD + d] = (__bf16)o;
    }
}

// ---------------- residual + layernorm (+ optional gelu), fp32/bf16 outs ----------------

__global__ __launch_bounds__(256) void resid_ln_kernel(
    const float* __restrict__ X, const float* __restrict__ Y,
    const float* __restrict__ g, const float* __restrict__ beta,
    float* __restrict__ outF, __bf16* __restrict__ outB, int n, int doGelu)
{
    __shared__ float sdata[256];
    const int row = blockIdx.x;
    const int cnt = n >> 8;                 // n/256, <= 8
    const float* xr = X + (size_t)row * n;
    const float* yr = Y ? Y + (size_t)row * n : nullptr;

    float v[8];
    float s = 0.0f;
    for (int i = 0; i < cnt; ++i) {
        const int c = threadIdx.x + i * 256;
        float t = xr[c] + (yr ? yr[c] : 0.0f);
        v[i] = t; s += t;
    }
    const float mean = block_reduce_sum(s, sdata) / (float)n;
    float q = 0.0f;
    for (int i = 0; i < cnt; ++i) { float d = v[i] - mean; q += d * d; }
    const float var = block_reduce_sum(q, sdata) / (float)n;
    const float rstd = rsqrtf(var + 1e-5f);

    for (int i = 0; i < cnt; ++i) {
        const int c = threadIdx.x + i * 256;
        float o = (v[i] - mean) * rstd * g[c] + beta[c];
        if (doGelu) o = gelu_exact(o);
        const size_t idx = (size_t)row * n + c;
        if (outF) outF[idx] = o;
        if (outB) outB[idx] = (__bf16)o;
    }
}

// ---------------- mean over tokens ----------------

__global__ __launch_bounds__(256) void token_mean_kernel(
    const float* __restrict__ X, float* __restrict__ outF, __bf16* __restrict__ outB)
{
    const int b = blockIdx.x;
    for (int hcol = threadIdx.x; hcol < HDIM; hcol += 256) {
        float s = 0.0f;
        for (int t = 0; t < SEQ; ++t) s += X[((size_t)b * SEQ + t) * HDIM + hcol];
        s *= (1.0f / SEQ);
        outF[(size_t)b * HDIM + hcol] = s;
        outB[(size_t)b * HDIM + hcol] = (__bf16)s;
    }
}

// ---------------- integration score ----------------

__global__ __launch_bounds__(256) void integration_kernel(
    const float* __restrict__ signals, const float* __restrict__ sal,
    float* __restrict__ out)
{
    __shared__ float sdata[256];
    const int b = blockIdx.x;
    const float* S = signals + (size_t)b * 4 * HDIM;

    float ps[4] = {0.f, 0.f, 0.f, 0.f};
    float pd[10] = {0.f, 0.f, 0.f, 0.f, 0.f, 0.f, 0.f, 0.f, 0.f, 0.f};
    for (int k = threadIdx.x; k < HDIM; k += 256) {
        const float a0 = S[k], a1 = S[HDIM + k], a2 = S[2 * HDIM + k], a3 = S[3 * HDIM + k];
        ps[0] += a0; ps[1] += a1; ps[2] += a2; ps[3] += a3;
        pd[0] += a0 * a0; pd[1] += a0 * a1; pd[2] += a0 * a2; pd[3] += a0 * a3;
        pd[4] += a1 * a1; pd[5] += a1 * a2; pd[6] += a1 * a3;
        pd[7] += a2 * a2; pd[8] += a2 * a3; pd[9] += a3 * a3;
    }
    float tot[14];
    for (int qi = 0; qi < 14; ++qi)
        tot[qi] = block_reduce_sum(qi < 4 ? ps[qi] : pd[qi - 4], sdata);

    if (threadIdx.x == 0) {
        const int pidx[4][4] = {{0, 1, 2, 3}, {1, 4, 5, 6}, {2, 5, 7, 8}, {3, 6, 8, 9}};
        float m[4];
        for (int i = 0; i < 4; ++i) m[i] = tot[i] / (float)HDIM;
        float acc = 0.0f;
        for (int i = 0; i < 4; ++i)
            for (int j = 0; j < 4; ++j) {
                float cov = tot[4 + pidx[i][j]] - (float)HDIM * m[i] * m[j];
                acc += fabsf(cov);
            }
        const float corr = acc * (1.0f / 16.0f);
        float sv[4], sm = 0.0f;
        for (int i = 0; i < 4; ++i) { sv[i] = sal[(size_t)b * 4 + i]; sm += sv[i]; }
        sm *= 0.25f;
        float var = 0.0f;
        for (int i = 0; i < 4; ++i) { float d = sv[i] - sm; var += d * d; }
        var *= (1.0f / 3.0f);                         // ddof=1
        out[b] = corr * (1.0f / (1.0f + expf(var)));  // sigmoid(-var)
    }
}

// ---------------- host orchestration ----------------

// d_in indices (setup_inputs dict order, depth-first):
//  0..3: world, self_sig, agency, meta
//  4..7: sal world w1,b1,w2,b2 | 8..11: sal self | 12..15: sal meta | 16: sal agency w
// 17..24: proj (world w,b)(self w,b)(agency w,b)(meta w,b)
// 25..88: 4 layers x [wq,bq,wk,bk,wv,bv,wo,bo,ln1g,ln1b,ffw1,ffb1,ffw2,ffb2,ln2g,ln2b]
// 89..94: dec w1,b1,ln_g,ln_b,w2,b2
// d_out: x[B,16,H] fp32 | integration[B] | broadcasted[B,H]

extern "C" void kernel_launch(void* const* d_in, const int* in_sizes, int n_in,
                              void* d_out, int out_size, void* d_ws, size_t ws_size,
                              hipStream_t stream) {
    (void)in_sizes; (void)n_in; (void)out_size; (void)ws_size;

    char* ws = (char*)d_ws;
    size_t off = 0;
    auto take = [&](size_t bytes) -> void* {
        void* p = ws + off;
        off += (bytes + 255) & ~(size_t)255;
        return p;
    };

    float*  signals = (float*)take((size_t)BATCH * 4 * HDIM * 4);
    float*  salbuf  = (float*)take((size_t)BATCH * 4 * 4);
    __bf16* xbf     = (__bf16*)take((size_t)TOKENS * HDIM * 2);
    float*  qkvf    = (float*)take((size_t)TOKENS * QKVN * 4);
    float*  qkvb    = (float*)take((size_t)4 * QKVN * 4);
    __bf16* obf     = (__bf16*)take((size_t)TOKENS * HDIM * 2);
    float*  y1      = (float*)take((size_t)TOKENS * HDIM * 4);
    __bf16* hbf     = (__bf16*)take((size_t)TOKENS * FFDIM * 2);
    float*  cons    = (float*)take((size_t)BATCH * HDIM * 4);
    __bf16* consbf  = (__bf16*)take((size_t)BATCH * HDIM * 2);
    float*  dy      = (float*)take((size_t)BATCH * 2 * HDIM * 4);
    __bf16* dybf    = (__bf16*)take((size_t)BATCH * 2 * HDIM * 2);

    const size_t M1 = 1048576;                    // 1024*1024 elems
    __bf16* wbf = (__bf16*)take((4 * 12 * M1 + 4 * M1) * 2);

    float* xout  = (float*)d_out;                          // [B,16,H]
    float* integ = xout + (size_t)TOKENS * HDIM;           // [B]
    float* bcast = integ + BATCH;                          // [B,H]

    auto conv = [&](const void* src, __bf16* dst, size_t n) {
        f2bf_kernel<<<(unsigned)((n + 255) / 256), 256, 0, stream>>>(
            (const float*)src, dst, n);
    };
    auto gemm = [&](const __bf16* A, const __bf16* W, const float* bias,
                    float* oF, __bf16* oB, int M, int N, int K, int act) {
        dim3 grid(N / 256, M / 64);
        gemm_bf16_wmma<<<grid, 256, 0, stream>>>(A, W, bias, oF, oB, M, N, K, act);
    };
    auto F = [&](int i) { return (const float*)d_in[i]; };

    // --- convert weights to bf16 (wq|wk|wv contiguous => fused QKV weight) ---
    __bf16 *wqkv[4], *wo[4], *w1[4], *w2[4];
    for (int l = 0; l < 4; ++l) {
        __bf16* base = wbf + (size_t)l * 12 * M1;
        wqkv[l] = base;                       // [3072,1024]
        wo[l] = base + 3 * M1;  w1[l] = base + 4 * M1;  w2[l] = base + 8 * M1;
        const int L = 25 + 16 * l;
        conv(d_in[L + 0], base, M1);            // wq
        conv(d_in[L + 2], base + M1, M1);       // wk
        conv(d_in[L + 4], base + 2 * M1, M1);   // wv
        conv(d_in[L + 6], wo[l], M1);
        conv(d_in[L + 10], w1[l], 4 * M1);
        conv(d_in[L + 12], w2[l], 4 * M1);
        pack3_kernel<<<QKVN / 256, 256, 0, stream>>>(
            F(L + 1), F(L + 3), F(L + 5), qkvb + (size_t)l * QKVN);
    }
    __bf16* decw1 = wbf + 48 * M1;
    __bf16* decw2 = decw1 + 2 * M1;
    conv(d_in[89], decw1, 2 * M1);
    conv(d_in[93], decw2, 2 * M1);

    // --- projections + salience ---
    proj_kernel<<<BATCH, 256, 0, stream>>>(F(0), F(17), F(18), signals, 256, 0); // world
    proj_kernel<<<BATCH, 256, 0, stream>>>(F(1), F(19), F(20), signals, 128, 1); // self
    proj_kernel<<<BATCH, 256, 0, stream>>>(F(2), F(21), F(22), signals,   1, 2); // agency
    proj_kernel<<<BATCH, 256, 0, stream>>>(F(3), F(23), F(24), signals, 512, 3); // meta

    sal_mlp_kernel<<<BATCH, 256, 0, stream>>>(F(0), F(4),  F(5),  F(6),  F(7),  salbuf, 256, 0);
    sal_mlp_kernel<<<BATCH, 256, 0, stream>>>(F(1), F(8),  F(9),  F(10), F(11), salbuf, 128, 1);
    sal_mlp_kernel<<<BATCH, 256, 0, stream>>>(F(3), F(12), F(13), F(14), F(15), salbuf, 512, 3);
    sal_agency_kernel<<<(BATCH + 255) / 256, 256, 0, stream>>>(F(2), F(16), salbuf, BATCH);

    integration_kernel<<<BATCH, 256, 0, stream>>>(signals, salbuf, integ);
    build_ws_kernel<<<BATCH, 256, 0, stream>>>(signals, salbuf, xout, xbf);

    // --- 4 transformer encoder layers ---
    for (int l = 0; l < 4; ++l) {
        const int L = 25 + 16 * l;
        gemm(xbf, wqkv[l], qkvb + (size_t)l * QKVN, qkvf, nullptr,
             TOKENS, QKVN, HDIM, 0);                                   // fused QKV
        attention_kernel<<<dim3(BATCH, NHEADS), 256, 0, stream>>>(qkvf, obf);
        gemm(obf, wo[l], F(L + 7), y1, nullptr, TOKENS, HDIM, HDIM, 0);
        resid_ln_kernel<<<TOKENS, 256, 0, stream>>>(xout, y1, F(L + 8), F(L + 9),
                                                    xout, xbf, HDIM, 0);
        gemm(xbf, w1[l], F(L + 11), nullptr, hbf, TOKENS, FFDIM, HDIM, 1); // relu, bf16
        gemm(hbf, w2[l], F(L + 13), y1, nullptr, TOKENS, HDIM, FFDIM, 0);
        resid_ln_kernel<<<TOKENS, 256, 0, stream>>>(xout, y1, F(L + 14), F(L + 15),
                                                    xout, xbf, HDIM, 0);
    }

    // --- decoder ---
    token_mean_kernel<<<BATCH, 256, 0, stream>>>(xout, cons, consbf);
    gemm(consbf, decw1, F(90), dy, nullptr, BATCH, 2 * HDIM, HDIM, 0);
    resid_ln_kernel<<<BATCH, 256, 0, stream>>>(dy, nullptr, F(91), F(92),
                                               nullptr, dybf, 2 * HDIM, 1); // LN + gelu -> bf16
    gemm(dybf, decw2, F(94), bcast, nullptr, BATCH, HDIM, 2 * HDIM, 0);
}
